// SyntaxTransferEncoder_70635032150607
// MI455X (gfx1250) — matrix operand order repaired and checked by reading, not compiled, not run on hardware
//
#include <hip/hip_runtime.h>

// ---------------------------------------------------------------------------
// TreeLSTM (4-ary, depth 8) for gfx1250.
// E=128, H=128, A=4, N_NODES=87381. Levels chained through d_out (parent h/c
// = 4 contiguous child rows). GEMMs on v_wmma_f32_16x16x32_bf16.
// Weights pre-packed to bf16 WMMA B fragments in d_ws (1 MiB, L2-resident).
// Workgroup: 512 threads = 16 waves, 32 nodes. Wave = one gate-group
// (8 N-tiles: i,o,u,fx,fh0..3) x one 16-row M-tile -> 64 acc VGPRs.
// This is the largest blocking that fits the wave32 VGPR budget without
// spills (verified: 2x/4x M-blocking both forced scratch spills).
// ---------------------------------------------------------------------------

typedef __attribute__((ext_vector_type(16))) __bf16 v16bf;
typedef __attribute__((ext_vector_type(8)))  float  v8f;

union ABFrag { uint4 u[2]; v16bf v; };

#define N_NODES 87381

// ws layout (bf16 elements):
#define OFF_WIOU 0        // KT=4,  NT=24  (49152)
#define OFF_WF   49152    // KT=4,  NT=8   (16384)
#define OFF_UIOU 65536    // KT=16, NT=24  (196608)
#define OFF_UF   262144   // KT=16, NT=32  (262144)

static __device__ inline unsigned int bf_bits(float f) {
  union { float f; unsigned int u; } x; x.f = f;
  return (x.u + 0x7FFFu + ((x.u >> 16) & 1u)) >> 16;  // RNE
}

static __device__ inline __bf16 f2bf(float f) {
  unsigned short s = (unsigned short)bf_bits(f);
  __bf16 b; __builtin_memcpy(&b, &s, 2); return b;
}

// two floats -> packed bf16x2 (lo = a, hi = b)
static __device__ inline unsigned int f2bf2(float a, float b) {
  return (bf_bits(a) & 0xFFFFu) | (bf_bits(b) << 16);
}

static __device__ inline float sigmoidf_(float x) {
  return 1.0f / (1.0f + __expf(-x));
}

static __device__ inline v8f wmma_bf16(v16bf a, v16bf b, v8f c) {
  return __builtin_amdgcn_wmma_f32_16x16x32_bf16(
      false, a, false, b, (short)0, c, false, false);
}

// B fragment: 32x16 (KxN) bf16, lane-major, 16 halfs/lane.
// Lane L<16: col N=L, K = kt*32 + 0..15 ; lane L>=16: col N=L-16, K=+16..31.
static __device__ inline v16bf load_b(const __bf16* __restrict__ base,
                                      int frag, int lane) {
  const uint4* q = (const uint4*)(base + ((size_t)frag << 9) + (lane << 4));
  ABFrag t; t.u[0] = q[0]; t.u[1] = q[1]; return t.v;
}

// A fragment: 16x32 (MxK) bf16 from LDS row-major (stride 648 halfs).
// Lane L<16: row M=L, K = cb + 0..15; lane L>=16: row M=L-16,
// K = cb + 16..31.  rowp points at As[row] + kb (kb = 0 or 16).
static __device__ inline v16bf load_a(const __bf16* rowp, int cb) {
  const uint4* p = (const uint4*)(rowp + cb);
  ABFrag t; t.u[0] = p[0]; t.u[1] = p[1]; return t.v;
}

// ---------------------------------------------------------------------------
// Weight pack: f32 (Nout x Korig row-major, used as B=W^T) -> bf16 fragments.
// ---------------------------------------------------------------------------
__global__ __launch_bounds__(256) void pack_bfrag(
    const float* __restrict__ src, int Korig, int KT, int totalElems,
    __bf16* __restrict__ dst) {
  int idx = blockIdx.x * 256 + threadIdx.x;
  if (idx >= totalElems) return;
  int j    = idx & 15;
  int lane = (idx >> 4) & 31;
  int frag = idx >> 9;
  int kt   = frag % KT;
  int nt   = frag / KT;
  int n = nt * 16 + (lane & 15);
  int k = kt * 32 + ((lane >> 4) << 4) + j;
  dst[idx] = f2bf(src[(size_t)n * Korig + k]);
}

// ---------------------------------------------------------------------------
// One tree level. 512 threads = 16 waves, 32 nodes per block.
// Wave w: gate-group g = w&7 (output cols [g*16,g*16+16) of i,o,u,fx,fh0..3),
// M-tile = w>>3 (rows mtile*16 .. +15). Exactly 128 K-steps per wave
// (20 x 3 iou + 4 fx + 64 fh) -> perfectly balanced waves.
// Epilogue is lane-local (no LDS round trip for accumulators).
// ---------------------------------------------------------------------------
__global__ __launch_bounds__(512) void tree_lstm_level(
    const float* __restrict__ emb,    // level-local: node g at emb[g*128]
    const float* __restrict__ hIn,    // bcast ? h0[512] : hIn[g*512 + ...]
    const float* __restrict__ cIn,    // bcast ? c0[512] : cIn[g*512 + ...]
    const __bf16* __restrict__ wfrag,
    const float* __restrict__ b_iou, const float* __restrict__ b_uiou,
    const float* __restrict__ b_wf,  const float* __restrict__ b_uf,
    float* __restrict__ outH, float* __restrict__ outC,
    int nNodes, int bcast) {
  __shared__ __align__(16) __bf16 As[32][648];  // 32 x 640 + pad (40.5 KB)

  const int tid      = threadIdx.x;
  const int tileBase = blockIdx.x * 32;

  // ---- Stage A = [x | h] as bf16 into LDS (packed pair stores) -----------
  for (int p = tid; p < 32 * 320; p += 512) {
    int r  = p / 320;
    int c2 = p - r * 320;      // pair index; cols c = 2*c2, 2*c2+1
    int c  = c2 * 2;
    int g  = tileBase + r;
    float v0 = 0.0f, v1 = 0.0f;
    if (g < nNodes) {
      if (c < 128) {
        const float2 t = *(const float2*)(emb + (size_t)g * 128 + c);
        v0 = t.x; v1 = t.y;
      } else if (bcast) {
        const float2 t = *(const float2*)(hIn + (c - 128));
        v0 = t.x; v1 = t.y;
      } else {
        const float2 t = *(const float2*)(hIn + (size_t)g * 512 + (c - 128));
        v0 = t.x; v1 = t.y;
      }
    }
    ((unsigned int*)&As[r][0])[c2] = f2bf2(v0, v1);
  }
  __syncthreads();

  // ---- GEMM phase ---------------------------------------------------------
  const int lane  = tid & 31;
  const int wave  = tid >> 5;
  const int gidx  = wave & 7;          // gate group (output column slice)
  const int mtile = wave >> 3;         // 0..1 -> rows mtile*16 .. +15

  const __bf16* wiou = wfrag + OFF_WIOU;
  const __bf16* wf   = wfrag + OFF_WF;
  const __bf16* uiou = wfrag + OFF_UIOU;
  const __bf16* uf   = wfrag + OFF_UF;

  const __bf16* rowp = &As[mtile * 16 + (lane & 15)][(lane >> 4) * 16];

  v8f z = {0.f, 0.f, 0.f, 0.f, 0.f, 0.f, 0.f, 0.f};
  v8f acc_i = z, acc_o = z, acc_u = z, acc_x = z;
  v8f fa[4] = {z, z, z, z};

  // x-driven part: K = 128 (A cols 0..127)
#pragma unroll
  for (int kt = 0; kt < 4; ++kt) {
    v16bf aF = load_a(rowp, kt * 32);
    acc_i = wmma_bf16(aF, load_b(wiou, ( 0 + gidx) * 4 + kt, lane), acc_i);
    acc_o = wmma_bf16(aF, load_b(wiou, ( 8 + gidx) * 4 + kt, lane), acc_o);
    acc_u = wmma_bf16(aF, load_b(wiou, (16 + gidx) * 4 + kt, lane), acc_u);
    acc_x = wmma_bf16(aF, load_b(wf,   gidx * 4 + kt,        lane), acc_x);
  }

  // h-driven part: K = 512 (A cols 128..639)
#pragma unroll 4
  for (int kt = 0; kt < 16; ++kt) {
    v16bf aF = load_a(rowp, 128 + kt * 32);
    acc_i = wmma_bf16(aF, load_b(uiou, ( 0 + gidx) * 16 + kt, lane), acc_i);
    acc_o = wmma_bf16(aF, load_b(uiou, ( 8 + gidx) * 16 + kt, lane), acc_o);
    acc_u = wmma_bf16(aF, load_b(uiou, (16 + gidx) * 16 + kt, lane), acc_u);
#pragma unroll
    for (int a = 0; a < 4; ++a)
      fa[a] = wmma_bf16(aF, load_b(uf, (a * 8 + gidx) * 16 + kt, lane), fa[a]);
  }

  // ---- Lane-local LSTM epilogue ------------------------------------------
  // C/D layout: lane L holds col N=L&15, rows M=(L>=16?8:0)+j (j = vgpr idx).
  const int hc = gidx * 16 + (lane & 15);
  const float bi_ = b_iou[hc]       + b_uiou[hc];
  const float bo_ = b_iou[128 + hc] + b_uiou[128 + hc];
  const float bu_ = b_iou[256 + hc] + b_uiou[256 + hc];
  const float bx_ = b_wf[hc];
  float bf4[4], cb4[4];
#pragma unroll
  for (int a = 0; a < 4; ++a) {
    bf4[a] = b_uf[a * 128 + hc];
    cb4[a] = bcast ? cIn[a * 128 + hc] : 0.0f;
  }
  const int mtop = (lane >> 4) * 8;

#pragma unroll
  for (int j = 0; j < 8; ++j) {
    int g = tileBase + mtile * 16 + mtop + j;
    if (g < nNodes) {
      float iv  = sigmoidf_(acc_i[j] + bi_);
      float ov  = sigmoidf_(acc_o[j] + bo_);
      float uv  = tanhf(acc_u[j] + bu_);
      float fxv = acc_x[j] + bx_;
      float csum = 0.0f;
#pragma unroll
      for (int a = 0; a < 4; ++a) {
        float fv = sigmoidf_(fa[a][j] + bf4[a] + fxv);
        float cv = bcast ? cb4[a] : cIn[(size_t)g * 512 + a * 128 + hc];
        csum += fv * cv;
      }
      float cn = iv * uv + csum;
      float hn = ov * tanhf(cn);
      outH[(size_t)g * 128 + hc] = hn;
      outC[(size_t)g * 128 + hc] = cn;
    }
  }
}

// ---------------------------------------------------------------------------
// Host launcher
// ---------------------------------------------------------------------------
extern "C" void kernel_launch(void* const* d_in, const int* in_sizes, int n_in,
                              void* d_out, int out_size, void* d_ws,
                              size_t ws_size, hipStream_t stream) {
  const float* emb    = (const float*)d_in[0];
  const float* W_iou  = (const float*)d_in[1];
  const float* b_iou  = (const float*)d_in[2];
  const float* U_iou  = (const float*)d_in[3];
  const float* b_uiou = (const float*)d_in[4];
  const float* W_f    = (const float*)d_in[5];
  const float* b_wf   = (const float*)d_in[6];
  const float* U_f    = (const float*)d_in[7];
  const float* b_uf   = (const float*)d_in[8];
  const float* h0     = (const float*)d_in[9];
  const float* c0     = (const float*)d_in[10];

  __bf16* wsb = (__bf16*)d_ws;  // needs 524288 * 2 = 1 MiB

  pack_bfrag<<<(49152  + 255) / 256, 256, 0, stream>>>(W_iou, 128,  4,  49152, wsb + OFF_WIOU);
  pack_bfrag<<<(16384  + 255) / 256, 256, 0, stream>>>(W_f,   128,  4,  16384, wsb + OFF_WF);
  pack_bfrag<<<(196608 + 255) / 256, 256, 0, stream>>>(U_iou, 512, 16, 196608, wsb + OFF_UIOU);
  pack_bfrag<<<(262144 + 255) / 256, 256, 0, stream>>>(U_f,   512, 16, 262144, wsb + OFF_UF);

  // Leaf-first level row bases (matches reference embedding/output layout).
  int lb[9];
  {
    int b = 0;
    for (int l = 8; l >= 0; --l) { lb[l] = b; b += 1 << (2 * l); }
  }
  float* outH = (float*)d_out;
  float* outC = outH + (size_t)N_NODES * 128;

  for (int l = 8; l >= 0; --l) {
    int n = 1 << (2 * l);
    const float* hI = (l == 8) ? h0 : outH + (size_t)lb[l + 1] * 128;
    const float* cI = (l == 8) ? c0 : outC + (size_t)lb[l + 1] * 128;
    int grid = (n + 31) / 32;
    tree_lstm_level<<<grid, 512, 0, stream>>>(
        emb + (size_t)lb[l] * 128, hI, cI, wsb,
        b_iou, b_uiou, b_wf, b_uf,
        outH + (size_t)lb[l] * 128, outC + (size_t)lb[l] * 128,
        n, (l == 8) ? 1 : 0);
  }
}
